// DynamicGraphConvolution_27075473834198
// MI455X (gfx1250) — matrix operand order repaired and checked by reading, not compiled, not run on hardware
//
#include <hip/hip_runtime.h>
#include <cstdint>

typedef __attribute__((ext_vector_type(4)))  float        v4f;
typedef __attribute__((ext_vector_type(8)))  float        v8f;
typedef __attribute__((ext_vector_type(4)))  unsigned int v4u;
typedef __attribute__((ext_vector_type(8)))  __bf16       v8bf;
typedef __attribute__((ext_vector_type(16))) __bf16       v16bf;
typedef int v4i __attribute__((vector_size(16)));   // gcc-style, for async builtin

#define DEV __device__ __forceinline__

constexpr int   Bk     = 256;
constexpr int   Nk     = 80;
constexpr int   Ck     = 1024;
constexpr int   COk    = 1024;
constexpr float SLOPEk = 0.2f;
constexpr float BNEPSk = 1e-5f;

DEV float leakyf(float v) { return v >= 0.f ? v : SLOPEk * v; }
DEV float sigmf(float v)  { return 1.f / (1.f + __expf(-v)); }

// ---------------------------------------------------------------------------
// 16-byte global -> LDS copy.  Uses CDNA5 async-to-LDS when the toolchain
// exposes it (tracked by ASYNCcnt), otherwise a synchronous VGPR bounce.
// ---------------------------------------------------------------------------
#if defined(__AMDGCN__) && __has_builtin(__builtin_amdgcn_global_load_async_to_lds_b128)
#define HAS_ASYNC_LDS 1
#else
#define HAS_ASYNC_LDS 0
#endif

DEV void async_cp16(const void* gsrc, void* ldst) {
#if HAS_ASYNC_LDS
  __builtin_amdgcn_global_load_async_to_lds_b128(
      (__attribute__((address_space(1))) v4i*)(uintptr_t)gsrc,
      (__attribute__((address_space(3))) v4i*)(uint32_t)(uintptr_t)ldst,
      0, 0);
#else
  *(v4u*)ldst = *(const v4u*)gsrc;
#endif
}

DEV void wait_async() {
#if HAS_ASYNC_LDS
#if __has_builtin(__builtin_amdgcn_s_wait_asynccnt)
  __builtin_amdgcn_s_wait_asynccnt(0);
#else
  asm volatile("s_wait_asynccnt 0x0" ::: "memory");
#endif
#endif
}

// ---------------------------------------------------------------------------
// bf16x3 WMMA GEMM with pre-split bf16 hi/lo operands on BOTH sides.
// C[M,Ncols] = epilogue( A[M,K] * Bt[Ncols,K]^T ),  full tiles only
// (M % 128 == 0, Ncols % 64 == 0).  All staging is 16B async-to-LDS copies,
// double-buffered: issue stage k+1, compute stage k, one barrier per K-step.
// EPI: 0 = +res (residual add), 1 = leaky.
// ---------------------------------------------------------------------------
template <int EPI>
__global__ __launch_bounds__(256) void wmma_gemm_pre(
    const __bf16* __restrict__ AH, const __bf16* __restrict__ AL, int lda,
    const __bf16* __restrict__ BH, const __bf16* __restrict__ BL, int ldb,
    float* __restrict__ Cg, int ldc,
    const float* __restrict__ res,
    int M, int Ncols, int K)
{
  // padded row stride of 40 bf16 (80B): conflict-free 16B LDS fragment reads
  __shared__ __bf16 sAhi[2][128 * 40];
  __shared__ __bf16 sAlo[2][128 * 40];
  __shared__ __bf16 sBhi[2][64 * 40];
  __shared__ __bf16 sBlo[2][64 * 40];

  const int tid  = threadIdx.x;
  const int lane = tid & 31;
  const int wave = tid >> 5;
  const int wM   = (wave & 3) * 32;   // 4 waves along M
  const int wN   = (wave >> 2) * 32;  // 2 waves along N
  const int Mb   = blockIdx.y * 128;
  const int Nb   = blockIdx.x * 64;

  // issue all async copies for one K-stage into LDS buffer `stg`
  auto issue = [&](int k0, int stg) {
#pragma unroll
    for (int i = 0; i < 2; ++i) {           // A: 512 16B chunks / 256 threads
      int linear = tid + i * 256;           // 0..511
      int row = linear >> 2;                // 0..127
      int q   = linear & 3;                 // 4 chunks of 8 bf16 per row
      long long g = (long long)(Mb + row) * lda + k0 + q * 8;
      async_cp16(AH + g, &sAhi[stg][row * 40 + q * 8]);
      async_cp16(AL + g, &sAlo[stg][row * 40 + q * 8]);
    }
    {                                        // B: 256 chunks / 256 threads
      int row = tid >> 2;                    // 0..63
      int q   = tid & 3;
      long long g = (long long)(Nb + row) * ldb + k0 + q * 8;
      async_cp16(BH + g, &sBhi[stg][row * 40 + q * 8]);
      async_cp16(BL + g, &sBlo[stg][row * 40 + q * 8]);
    }
  };

  v8f acc[2][2];
  const v8f vz = {0.f, 0.f, 0.f, 0.f, 0.f, 0.f, 0.f, 0.f};
  acc[0][0] = vz; acc[0][1] = vz; acc[1][0] = vz; acc[1][1] = vz;

  const int kbA = (lane >> 4) * 8;   // A frag: K 0..7 / 8..15 (+16 for upper)
  const int kbB = (lane >> 4) * 8;   // B frag: K 0..7 / 8..15 (+16 for upper)
  const int fm  = lane & 15;

  issue(0, 0);
  int st = 0;
  for (int k0 = 0; k0 < K; k0 += 32) {
    wait_async();        // this wave's copies for stage `st` have landed
    __syncthreads();     // all waves' copies landed; prev compute done
    if (k0 + 32 < K) issue(k0 + 32, st ^ 1);

    v16bf aHi[2], aLo[2], bHi[2], bLo[2];
#pragma unroll
    for (int mi = 0; mi < 2; ++mi) {
      int base = (wM + mi * 16 + fm) * 40 + kbA;
      v8bf h0 = *(const v8bf*)(&sAhi[st][base]);
      v8bf h1 = *(const v8bf*)(&sAhi[st][base + 16]);
      v8bf l0 = *(const v8bf*)(&sAlo[st][base]);
      v8bf l1 = *(const v8bf*)(&sAlo[st][base + 16]);
      aHi[mi] = __builtin_shufflevector(h0, h1, 0,1,2,3,4,5,6,7,8,9,10,11,12,13,14,15);
      aLo[mi] = __builtin_shufflevector(l0, l1, 0,1,2,3,4,5,6,7,8,9,10,11,12,13,14,15);
    }
#pragma unroll
    for (int ni = 0; ni < 2; ++ni) {
      int base = (wN + ni * 16 + fm) * 40 + kbB;
      v8bf h0 = *(const v8bf*)(&sBhi[st][base]);
      v8bf h1 = *(const v8bf*)(&sBhi[st][base + 16]);
      v8bf l0 = *(const v8bf*)(&sBlo[st][base]);
      v8bf l1 = *(const v8bf*)(&sBlo[st][base + 16]);
      bHi[ni] = __builtin_shufflevector(h0, h1, 0,1,2,3,4,5,6,7,8,9,10,11,12,13,14,15);
      bLo[ni] = __builtin_shufflevector(l0, l1, 0,1,2,3,4,5,6,7,8,9,10,11,12,13,14,15);
    }
    // bf16x3: hi*hi + hi*lo + lo*hi
#pragma unroll
    for (int mi = 0; mi < 2; ++mi)
#pragma unroll
      for (int ni = 0; ni < 2; ++ni) {
        acc[mi][ni] = __builtin_amdgcn_wmma_f32_16x16x32_bf16(
            false, aHi[mi], false, bHi[ni], (short)0, acc[mi][ni], false, false);
        acc[mi][ni] = __builtin_amdgcn_wmma_f32_16x16x32_bf16(
            false, aHi[mi], false, bLo[ni], (short)0, acc[mi][ni], false, false);
        acc[mi][ni] = __builtin_amdgcn_wmma_f32_16x16x32_bf16(
            false, aLo[mi], false, bHi[ni], (short)0, acc[mi][ni], false, false);
      }
    st ^= 1;
  }

  // epilogue; C layout: VGPR r -> row r + 8*(lane>>4), col = lane&15
  const int hi = lane >> 4;
#pragma unroll
  for (int mi = 0; mi < 2; ++mi)
#pragma unroll
    for (int ni = 0; ni < 2; ++ni)
#pragma unroll
      for (int r = 0; r < 8; ++r) {
        int row = Mb + wM + mi * 16 + r + hi * 8;
        int col = Nb + wN + ni * 16 + fm;
        float v = acc[mi][ni][r];
        if (EPI == 0)      v += res[(long long)row * ldc + col];
        else if (EPI == 1) v = leakyf(v);
        Cg[(long long)row * ldc + col] = v;
      }
}

// ---------------------------------------------------------------------------
// bf16x3 WMMA GEMM, fp32 operands split hi/lo on the fly (handles partial
// tiles and a batch dimension).  Same tiling as above, single-buffered.
// EPI: 2 = +bias[col], 3 = sigmoid(acc + rowAux[z][row] + bias[row])
// ---------------------------------------------------------------------------
template <int EPI>
__global__ __launch_bounds__(256) void wmma_gemm_f32(
    const float* __restrict__ Ag, int lda, long long strideA,
    const float* __restrict__ BtF, int ldb, long long strideB,
    float* __restrict__ Cg, int ldc, long long strideC,
    const float* __restrict__ rowAux, int auxStride,
    const float* __restrict__ biasv,
    int M, int Ncols, int K)
{
  __shared__ __bf16 sAhi[128 * 40];
  __shared__ __bf16 sAlo[128 * 40];
  __shared__ __bf16 sBhi[64 * 40];
  __shared__ __bf16 sBlo[64 * 40];

  const int tid  = threadIdx.x;
  const int lane = tid & 31;
  const int wave = tid >> 5;
  const int wM   = (wave & 3) * 32;
  const int wN   = (wave >> 2) * 32;
  const int z    = blockIdx.z;
  const int Mb   = blockIdx.y * 128;
  const int Nb   = blockIdx.x * 64;

  const float* Ab = Ag + (long long)z * strideA;
  const float* Bb = BtF + (long long)z * strideB;
  float*       Cb = Cg + (long long)z * strideC;

  v8f acc[2][2];
  const v8f vz = {0.f, 0.f, 0.f, 0.f, 0.f, 0.f, 0.f, 0.f};
  acc[0][0] = vz; acc[0][1] = vz; acc[1][0] = vz; acc[1][1] = vz;

  const int kbA = (lane >> 4) * 8;
  const int kbB = (lane >> 4) * 8;
  const int fm  = lane & 15;

  for (int k0 = 0; k0 < K; k0 += 32) {
    // stage A tile: 128 x 32 fp32 -> bf16 hi/lo
#pragma unroll
    for (int i = 0; i < 4; ++i) {
      int linear = tid + i * 256;
      int row = linear >> 3;
      int c4  = linear & 7;
      v4f v = {0.f, 0.f, 0.f, 0.f};
      int gr = Mb + row;
      if (gr < M) v = *(const v4f*)(Ab + (long long)gr * lda + k0 + c4 * 4);
#pragma unroll
      for (int j = 0; j < 4; ++j) {
        float f  = v[j];
        __bf16 h = (__bf16)f;
        sAhi[row * 40 + c4 * 4 + j] = h;
        sAlo[row * 40 + c4 * 4 + j] = (__bf16)(f - (float)h);
      }
    }
    // stage B tile: 64 x 32 fp32 -> bf16 hi/lo
#pragma unroll
    for (int i = 0; i < 2; ++i) {
      int linear = tid + i * 256;
      int n  = linear >> 3;
      int c4 = linear & 7;
      v4f v = {0.f, 0.f, 0.f, 0.f};
      int gn = Nb + n;
      if (gn < Ncols) v = *(const v4f*)(Bb + (long long)gn * ldb + k0 + c4 * 4);
#pragma unroll
      for (int j = 0; j < 4; ++j) {
        float f  = v[j];
        __bf16 h = (__bf16)f;
        sBhi[n * 40 + c4 * 4 + j] = h;
        sBlo[n * 40 + c4 * 4 + j] = (__bf16)(f - (float)h);
      }
    }
    if (k0 + 32 < K) {
      int gr = Mb + (tid >> 1);
      if (gr < M) __builtin_prefetch(Ab + (long long)gr * lda + (k0 + 32), 0, 1);
    }
    __syncthreads();

    v16bf aHi[2], aLo[2], bHi[2], bLo[2];
#pragma unroll
    for (int mi = 0; mi < 2; ++mi) {
      int base = (wM + mi * 16 + fm) * 40 + kbA;
      v8bf h0 = *(const v8bf*)(&sAhi[base]);
      v8bf h1 = *(const v8bf*)(&sAhi[base + 16]);
      v8bf l0 = *(const v8bf*)(&sAlo[base]);
      v8bf l1 = *(const v8bf*)(&sAlo[base + 16]);
      aHi[mi] = __builtin_shufflevector(h0, h1, 0,1,2,3,4,5,6,7,8,9,10,11,12,13,14,15);
      aLo[mi] = __builtin_shufflevector(l0, l1, 0,1,2,3,4,5,6,7,8,9,10,11,12,13,14,15);
    }
#pragma unroll
    for (int ni = 0; ni < 2; ++ni) {
      int base = (wN + ni * 16 + fm) * 40 + kbB;
      v8bf h0 = *(const v8bf*)(&sBhi[base]);
      v8bf h1 = *(const v8bf*)(&sBhi[base + 16]);
      v8bf l0 = *(const v8bf*)(&sBlo[base]);
      v8bf l1 = *(const v8bf*)(&sBlo[base + 16]);
      bHi[ni] = __builtin_shufflevector(h0, h1, 0,1,2,3,4,5,6,7,8,9,10,11,12,13,14,15);
      bLo[ni] = __builtin_shufflevector(l0, l1, 0,1,2,3,4,5,6,7,8,9,10,11,12,13,14,15);
    }
#pragma unroll
    for (int mi = 0; mi < 2; ++mi)
#pragma unroll
      for (int ni = 0; ni < 2; ++ni) {
        acc[mi][ni] = __builtin_amdgcn_wmma_f32_16x16x32_bf16(
            false, aHi[mi], false, bHi[ni], (short)0, acc[mi][ni], false, false);
        acc[mi][ni] = __builtin_amdgcn_wmma_f32_16x16x32_bf16(
            false, aHi[mi], false, bLo[ni], (short)0, acc[mi][ni], false, false);
        acc[mi][ni] = __builtin_amdgcn_wmma_f32_16x16x32_bf16(
            false, aLo[mi], false, bHi[ni], (short)0, acc[mi][ni], false, false);
      }
    __syncthreads();
  }

  const int hi = lane >> 4;
#pragma unroll
  for (int mi = 0; mi < 2; ++mi)
#pragma unroll
    for (int ni = 0; ni < 2; ++ni)
#pragma unroll
      for (int r = 0; r < 8; ++r) {
        int row = Mb + wM + mi * 16 + r + hi * 8;
        int col = Nb + wN + ni * 16 + fm;
        if (row < M && col < Ncols) {
          float v = acc[mi][ni][r];
          if (EPI == 2)      v += biasv[col];
          else if (EPI == 3) v = sigmf(v + rowAux[(long long)z * auxStride + row] + biasv[row]);
          Cb[(long long)row * ldc + col] = v;
        }
      }
}

// ---------------------------------------------------------------------------
// adj_s = D^-1/2 A^T D^-1/2
// ---------------------------------------------------------------------------
__global__ __launch_bounds__(128) void k_adj_s(const float* __restrict__ A,
                                               float* __restrict__ adjS) {
  __shared__ float sd[Nk];
  int tid = threadIdx.x;
  if (tid < Nk) {
    float s = 0.f;
    for (int j = 0; j < Nk; ++j) s += A[tid * Nk + j];
    sd[tid] = rsqrtf(s);
  }
  __syncthreads();
  for (int idx = tid; idx < Nk * Nk; idx += 128) {
    int i = idx / Nk, j = idx % Nk;
    adjS[idx] = sd[i] * A[j * Nk + i] * sd[j];
  }
}

// W[K][Ncols] fp32 -> Wt[n][k] bf16 hi/lo (transpose + split)
__global__ __launch_bounds__(256) void k_convert_wT(const float* __restrict__ W,
                                                    __bf16* __restrict__ Whi,
                                                    __bf16* __restrict__ Wlo,
                                                    int K, int Ncols) {
  long long idx = (long long)blockIdx.x * 256 + threadIdx.x;
  if (idx >= (long long)K * Ncols) return;
  int n = (int)(idx / K);
  int k = (int)(idx % K);
  float  f = W[(long long)k * Ncols + n];
  __bf16 h = (__bf16)f;
  Whi[idx] = h;
  Wlo[idx] = (__bf16)(f - (float)h);
}

// h[b] = leaky(adj @ x[b]), written as bf16 hi/lo pair (feeds the big GEMMs).
// adjStride=0 -> shared static adjacency.
__global__ __launch_bounds__(256) void k_gcn(const float* __restrict__ adj,
                                             long long adjStride,
                                             const float* __restrict__ xin,
                                             __bf16* __restrict__ hH,
                                             __bf16* __restrict__ hL) {
  __shared__ float sAdj[Nk * Nk];   // 25.6 KB
  __shared__ float sX[Nk * 68];     // 80 x 64 fp32 tile, padded
  int tid = threadIdx.x;
  int b   = blockIdx.y;
  int c0  = blockIdx.x * 64;
  const float* ab = adj + (long long)b * adjStride;
  for (int i = tid; i < (Nk * Nk) / 4; i += 256)        // 1600 16B chunks
    async_cp16(ab + i * 4, &sAdj[i * 4]);
  for (int i = tid; i < Nk * 16; i += 256) {            // 80 rows * 16 chunks
    int row = i >> 4;
    int c4  = i & 15;
    async_cp16(xin + (long long)b * Nk * Ck + (long long)row * Ck + c0 + c4 * 4,
               &sX[row * 68 + c4 * 4]);
  }
  wait_async();
  __syncthreads();
  int c = tid & 63;
  int g = tid >> 6;  // 0..3, 20 rows each
  float accv[20];
#pragma unroll
  for (int r = 0; r < 20; ++r) accv[r] = 0.f;
  for (int j = 0; j < Nk; ++j) {
    float xv = sX[j * 68 + c];
#pragma unroll
    for (int r = 0; r < 20; ++r) accv[r] += sAdj[(g * 20 + r) * Nk + j] * xv;
  }
#pragma unroll
  for (int r = 0; r < 20; ++r) {
    int row = g * 20 + r;
    long long idx = (long long)b * Nk * Ck + (long long)row * Ck + c0 + c;
    float  v = leakyf(accv[r]);
    __bf16 h = (__bf16)v;
    hH[idx] = h;
    hL[idx] = (__bf16)(v - (float)h);
  }
}

// glb0[b][c] = mean_n x2[b][n][c]
__global__ __launch_bounds__(256) void k_colmean(const float* __restrict__ x2,
                                                 float* __restrict__ glb0) {
  int c = blockIdx.x * 256 + threadIdx.x;
  int b = blockIdx.y;
  float s = 0.f;
  for (int n = 0; n < Nk; ++n) s += x2[(long long)b * Nk * Ck + (long long)n * Ck + c];
  glb0[(long long)b * Ck + c] = s * (1.f / Nk);
}

__global__ __launch_bounds__(256) void k_bn_stats(const float* __restrict__ glb1,
                                                  float* __restrict__ mu,
                                                  float* __restrict__ rstd) {
  int c = blockIdx.x * 256 + threadIdx.x;
  float s = 0.f, s2 = 0.f;
  for (int b = 0; b < Bk; ++b) {
    float v = glb1[(long long)b * Ck + c];
    s += v; s2 += v * v;
  }
  float m   = s * (1.f / Bk);
  float var = s2 * (1.f / Bk) - m * m;
  mu[c]   = m;
  rstd[c] = rsqrtf(var + BNEPSk);
}

__global__ __launch_bounds__(256) void k_bn_apply(const float* __restrict__ glb1,
                                                  const float* __restrict__ mu,
                                                  const float* __restrict__ rstd,
                                                  const float* __restrict__ gam,
                                                  const float* __restrict__ bet,
                                                  float* __restrict__ glb2) {
  long long idx = (long long)blockIdx.x * 256 + threadIdx.x;
  int c = (int)(idx & (Ck - 1));
  glb2[idx] = leakyf((glb1[idx] - mu[c]) * rstd[c] * gam[c] + bet[c]);
}

// s1[b][n] = sum_c Wco[n][c] * glb2[b][c]   (first C columns of conv_co_w)
__global__ __launch_bounds__(128) void k_s1(const float* __restrict__ glb2,
                                            const float* __restrict__ Wco,
                                            float* __restrict__ s1) {
  __shared__ float sg[Ck];
  int tid = threadIdx.x;
  int b   = blockIdx.x;
  for (int i = tid; i < Ck; i += 128) sg[i] = glb2[(long long)b * Ck + i];
  __syncthreads();
  if (tid < Nk) {
    float s = 0.f;
    for (int c = 0; c < Ck; ++c) s += Wco[(long long)tid * (2 * Ck) + c] * sg[c];
    s1[b * Nk + tid] = s;
  }
}

// per-batch loss partial: || sigmoid(out1[b]) - t[b] ||_2
__global__ __launch_bounds__(128) void k_loss_part(const float* __restrict__ out1,
                                                   const float* __restrict__ dyn,
                                                   float* __restrict__ part) {
  __shared__ float so[Nk];
  __shared__ float red[128];
  int tid = threadIdx.x;
  int b   = blockIdx.x;
  if (tid < Nk) so[tid] = sigmf(out1[b * Nk + tid]);
  __syncthreads();
  float sq = 0.f;
  if (tid < Nk) {
    float t = 0.f;
    for (int nn = 0; nn < Nk; ++nn) t += so[nn] * dyn[(long long)b * Nk * Nk + nn * Nk + tid];
    t *= (1.f / Nk);
    float d = so[tid] - t;
    sq = d * d;
  }
  red[tid] = sq;
  __syncthreads();
  for (int s = 64; s > 0; s >>= 1) {
    if (tid < s) red[tid] += red[tid + s];
    __syncthreads();
  }
  if (tid == 0) part[b] = sqrtf(red[0]);
}

__global__ __launch_bounds__(256) void k_loss_sum(const float* __restrict__ part,
                                                  float* __restrict__ dst) {
  __shared__ float red[256];
  int tid = threadIdx.x;
  red[tid] = part[tid];
  __syncthreads();
  for (int s = 128; s > 0; s >>= 1) {
    if (tid < s) red[tid] += red[tid + s];
    __syncthreads();
  }
  if (tid == 0) dst[0] = red[0];
}

// adjs[b] = D^-1/2 dyn[b]^T D^-1/2
__global__ __launch_bounds__(256) void k_gen_adjs(const float* __restrict__ dyn,
                                                  float* __restrict__ adjs) {
  __shared__ float sD[Nk * Nk];
  __shared__ float sd[Nk];
  int tid = threadIdx.x;
  int b   = blockIdx.x;
  for (int i = tid; i < (Nk * Nk) / 4; i += 256)
    async_cp16(dyn + (long long)b * Nk * Nk + i * 4, &sD[i * 4]);
  wait_async();
  __syncthreads();
  if (tid < Nk) {
    float s = 0.f;
    for (int m = 0; m < Nk; ++m) s += sD[tid * Nk + m];
    sd[tid] = rsqrtf(s);
  }
  __syncthreads();
  for (int i = tid; i < Nk * Nk; i += 256) {
    int n = i / Nk, m = i % Nk;
    adjs[(long long)b * Nk * Nk + i] = sd[n] * sD[m * Nk + n] * sd[m];
  }
}

// ---------------------------------------------------------------------------
extern "C" void kernel_launch(void* const* d_in, const int* in_sizes, int n_in,
                              void* d_out, int out_size, void* d_ws, size_t ws_size,
                              hipStream_t stream) {
  (void)in_sizes; (void)n_in; (void)out_size; (void)ws_size;
  const float* x    = (const float*)d_in[0];
  const float* out1 = (const float*)d_in[1];
  const float* A    = (const float*)d_in[2];
  const float* Wst  = (const float*)d_in[3];
  const float* Wg   = (const float*)d_in[4];
  const float* bg   = (const float*)d_in[5];
  const float* gam  = (const float*)d_in[6];
  const float* bet  = (const float*)d_in[7];
  const float* Wco  = (const float*)d_in[8];
  const float* bco  = (const float*)d_in[9];
  const float* Wdy  = (const float*)d_in[10];
  float* out = (float*)d_out;

  char*  base = (char*)d_ws;
  size_t off  = 0;
  auto wsAlloc = [&](size_t bytes) -> void* {
    void* p = base + off;
    off += (bytes + 255) & ~(size_t)255;
    return p;
  };
  float*  adjS  = (float*)wsAlloc((size_t)Nk * Nk * 4);
  __bf16* hH    = (__bf16*)wsAlloc((size_t)Bk * Nk * Ck * 2);  // h1 then h2 (hi)
  __bf16* hL    = (__bf16*)wsAlloc((size_t)Bk * Nk * Ck * 2);  // h1 then h2 (lo)
  float*  x2    = (float*)wsAlloc((size_t)Bk * Nk * Ck * 4);
  __bf16* wstH  = (__bf16*)wsAlloc((size_t)Ck * Ck * 2);
  __bf16* wstL  = (__bf16*)wsAlloc((size_t)Ck * Ck * 2);
  __bf16* wdyH  = (__bf16*)wsAlloc((size_t)Ck * COk * 2);
  __bf16* wdyL  = (__bf16*)wsAlloc((size_t)Ck * COk * 2);
  float*  glb0  = (float*)wsAlloc((size_t)Bk * Ck * 4);
  float*  glb1  = (float*)wsAlloc((size_t)Bk * Ck * 4);
  float*  glb2  = (float*)wsAlloc((size_t)Bk * Ck * 4);
  float*  muv   = (float*)wsAlloc((size_t)Ck * 4);
  float*  rstdv = (float*)wsAlloc((size_t)Ck * 4);
  float*  s1v   = (float*)wsAlloc((size_t)Bk * Nk * 4);
  float*  dynv  = (float*)wsAlloc((size_t)Bk * Nk * Nk * 4);
  float*  adjsv = (float*)wsAlloc((size_t)Bk * Nk * Nk * 4);
  float*  partv = (float*)wsAlloc((size_t)Bk * 4);

  // static adjacency + weight conversion (bf16 hi/lo, transposed)
  k_adj_s<<<1, 128, 0, stream>>>(A, adjS);
  {
    int blocks = (int)(((long long)Ck * Ck + 255) / 256);
    k_convert_wT<<<blocks, 256, 0, stream>>>(Wst, wstH, wstL, Ck, Ck);
    k_convert_wT<<<blocks, 256, 0, stream>>>(Wdy, wdyH, wdyL, Ck, COk);
  }
  // h1 = leaky(adj_s @ x)  -> bf16 hi/lo
  k_gcn<<<dim3(Ck / 64, Bk), 256, 0, stream>>>(adjS, 0, x, hH, hL);
  // x2 = x + h1 @ static_weight   (async double-buffered WMMA GEMM)
  wmma_gemm_pre<0><<<dim3(Ck / 64, (Bk * Nk) / 128), 256, 0, stream>>>(
      hH, hL, Ck, wstH, wstL, Ck, x2, Ck, x, Bk * Nk, Ck, Ck);
  // global pooling + 1x1 conv + BN + leaky
  k_colmean<<<dim3(Ck / 256, Bk), 256, 0, stream>>>(x2, glb0);
  wmma_gemm_f32<2><<<dim3(Ck / 64, (Bk + 127) / 128, 1), 256, 0, stream>>>(
      glb0, Ck, 0, Wg, Ck, 0, glb1, Ck, 0, nullptr, 0, bg, Bk, Ck, Ck);
  k_bn_stats<<<Ck / 256, 256, 0, stream>>>(glb1, muv, rstdv);
  k_bn_apply<<<(Bk * Ck) / 256, 256, 0, stream>>>(glb1, muv, rstdv, gam, bet, glb2);
  // dyn = sigmoid(Wco_a @ glb2 + Wco_b @ x2^T + bco)
  k_s1<<<Bk, 128, 0, stream>>>(glb2, Wco, s1v);
  wmma_gemm_f32<3><<<dim3((Nk + 63) / 64, (Nk + 127) / 128, Bk), 256, 0, stream>>>(
      Wco + Ck, 2 * Ck, 0, x2, Ck, (long long)Nk * Ck,
      dynv, Nk, (long long)Nk * Nk, s1v, Nk, bco, Nk, Nk, Ck);
  // adjacency consistency loss (deterministic two-stage reduction)
  k_loss_part<<<Bk, 128, 0, stream>>>(out1, dynv, partv);
  k_loss_sum<<<1, 256, 0, stream>>>(partv, out + (size_t)Bk * Nk * COk);
  // dynamic GCN
  k_gen_adjs<<<Bk, 256, 0, stream>>>(dynv, adjsv);
  k_gcn<<<dim3(Ck / 64, Bk), 256, 0, stream>>>(adjsv, (long long)Nk * Nk, x2, hH, hL);
  // out = leaky(h2 @ dynamic_weight)
  wmma_gemm_pre<1><<<dim3(COk / 64, (Bk * Nk) / 128), 256, 0, stream>>>(
      hH, hL, Ck, wdyH, wdyL, Ck, out, COk, nullptr, Bk * Nk, COk, Ck);
}
